// Attention_64690797412805
// MI455X (gfx1250) — matrix operand order repaired and checked by reading, not compiled, not run on hardware
//
#include <hip/hip_runtime.h>
#include <hip/hip_bf16.h>

// ---------------------------------------------------------------------------
// MI455X (gfx1250) attention block: QKV proj (+fused RoPE) -> flash attention
// -> output proj. All GEMMs via v_wmma_f32_16x16x32_bf16 (wave32).
// GEMM tiles stream through the CDNA5 async LDS-DMA path
// (global_load_async_to_lds_b128 + s_wait_asynccnt) with double buffering.
// ---------------------------------------------------------------------------

typedef __bf16 bf16;
typedef __bf16 v16bf __attribute__((ext_vector_type(16)));
typedef __bf16 bf16x2 __attribute__((ext_vector_type(2)));
typedef float  v8f   __attribute__((ext_vector_type(8)));

#define Bc   2
#define Sc   2048
#define DIMc 4096
#define Hc   32
#define KVc  8
#define HDc  128
#define REPc (Hc / KVc)
#define NQc  (Hc * HDc)    // 4096
#define NKVc (KVc * HDc)   // 1024
#define Mc   (Bc * Sc)     // 4096
#define SCALEc 0.08838834764831845f  // 128^-0.5
#define LDP  40            // padded LDS row stride (16B-aligned, conflict-free)

// A-fragment (16x32 bf16) K index for vgpr-pair j, lane half h  (ISA 7.12.2)
__device__ __forceinline__ int akidx(int j, int h) {
    return ((j >> 2) << 4) + h * 8 + ((j & 3) << 1);
}
// B-fragment (32x16 bf16) K index (same K order as A fragment)
__device__ __forceinline__ int bkidx(int j, int h) {
    return ((j >> 2) << 4) + h * 8 + ((j & 3) << 1);
}

__device__ __forceinline__ v8f wmma_bf16(v16bf a, v16bf b, v8f c) {
    return __builtin_amdgcn_wmma_f32_16x16x32_bf16(false, a, false, b,
                                                   (short)0, c, false, false);
}

// CDNA5 async DMA: 16B per lane, global -> LDS, tracked with ASYNCcnt.
// Generic LDS pointers carry the LDS byte offset in addr[31:0] (ISA 10.2),
// so the VDST operand is just the low 32 bits of the generic pointer.
__device__ __forceinline__ void async_lds_b128(const void* g, void* l) {
    unsigned loff = (unsigned)(size_t)l;
    unsigned long long ga = (unsigned long long)(size_t)g;
    asm volatile("global_load_async_to_lds_b128 %0, %1, off"
                 :: "v"(loff), "v"(ga) : "memory");
}

// ---------------------------------------------------------------------------
// f32 -> bf16 conversion (grid-stride) for activations
// ---------------------------------------------------------------------------
__global__ void cvt_bf16_k(const float* __restrict__ src, bf16* __restrict__ dst, int n) {
    int i = blockIdx.x * blockDim.x + threadIdx.x;
    int stride = gridDim.x * blockDim.x;
    for (; i < n; i += stride) dst[i] = (bf16)src[i];
}

// ---------------------------------------------------------------------------
// f32 -> bf16 conversion with transpose (for weights): src K x N  ->  dst N x K
// ---------------------------------------------------------------------------
__launch_bounds__(256)
__global__ void cvt_bf16_t_k(const float* __restrict__ src, bf16* __restrict__ dst,
                             int K, int N) {
    __shared__ float tile[32][33];
    const int t  = threadIdx.x;
    const int tx = t & 31, ty = t >> 5;          // 32 x 8
    const int nb = blockIdx.x * 32, kb = blockIdx.y * 32;
    for (int i = 0; i < 4; ++i) {
        int r = ty + 8 * i;
        tile[r][tx] = src[(size_t)(kb + r) * N + nb + tx];
    }
    __syncthreads();
    for (int i = 0; i < 4; ++i) {
        int r = ty + 8 * i;
        dst[(size_t)(nb + r) * K + kb + tx] = (bf16)tile[tx][r];
    }
}

// ---------------------------------------------------------------------------
// 128x128-tile WMMA GEMM:  C(MxN) = A(MxK,bf16) * Bt(NxK,bf16)^T
//   ROPE=1      : rotary embedding fused into epilogue
//   STOREMODE   : 0 = bf16 row-major, 1 = f32 row-major (d_out),
//                 2 = bf16 transposed V layout [b][g][d][s]
// 256 threads = 8 waves (2x4); each wave computes a 64x32 region (4x2 tiles).
// Double-buffered LDS tiles filled by async LDS-DMA; compute overlaps copy.
// ---------------------------------------------------------------------------
template<int ROPE, int STOREMODE>
__launch_bounds__(256)
__global__ void gemm_bf16_wmma_k(const bf16* __restrict__ A, const bf16* __restrict__ Bt,
                                 void* __restrict__ Cout, int M, int N, int K,
                                 const float* __restrict__ cosT,
                                 const float* __restrict__ sinT)
{
    __shared__ bf16 lA[2][128 * LDP];
    __shared__ bf16 lB[2][128 * LDP];

    const int t    = threadIdx.x;
    const int lane = t & 31;
    const int w    = t >> 5;
    const int wr   = w >> 2;     // 0..1
    const int wc   = w & 3;      // 0..3
    const int half = lane >> 4;
    const int ln   = lane & 15;
    const int blockM = blockIdx.y * 128;
    const int blockN = blockIdx.x * 128;

    const int ldRow = t >> 2;            // 0..63 (two passes -> 128 rows)
    const int ldCol = (t & 3) << 3;      // 0,8,16,24

    v8f acc[4][2];
    for (int mt = 0; mt < 4; ++mt)
        for (int nt = 0; nt < 2; ++nt)
            acc[mt][nt] = {};

    // Issue 4 async 16B transfers per thread for one K-tile (2 rows x {A,B}).
    auto issueTile = [&](int buf, int k0) {
        for (int p = 0; p < 2; ++p) {
            int row = ldRow + p * 64;
            async_lds_b128(&A[(size_t)(blockM + row) * K + k0 + ldCol],
                           &lA[buf][row * LDP + ldCol]);
            async_lds_b128(&Bt[(size_t)(blockN + row) * K + k0 + ldCol],
                           &lB[buf][row * LDP + ldCol]);
        }
    };

    issueTile(0, 0);

    for (int k0 = 0; k0 < K; k0 += 32) {
        const int  cur  = (k0 >> 5) & 1;
        const bool more = (k0 + 32) < K;
        if (more) issueTile(cur ^ 1, k0 + 32);
        // L2 prefetch two tiles ahead (async pipeline hides one tile).
        if (k0 + 64 < K) {
            __builtin_prefetch(&A[(size_t)(blockM + ldRow) * K + k0 + 64 + ldCol], 0, 0);
            __builtin_prefetch(&Bt[(size_t)(blockN + ldRow) * K + k0 + 64 + ldCol], 0, 0);
        }
        // Wait for the current tile (leave the 4 just-issued in flight).
        if (more) asm volatile("s_wait_asynccnt 0x4" ::: "memory");
        else      asm volatile("s_wait_asynccnt 0x0" ::: "memory");
        __syncthreads();

        const bf16* __restrict__ cA = lA[cur];
        const bf16* __restrict__ cB = lB[cur];

        v16bf af[4];
        for (int mt = 0; mt < 4; ++mt) {
            int row = wr * 64 + mt * 16 + ln;
            for (int j = 0; j < 8; ++j) {
                int kk = akidx(j, half);
                bf16x2 p = *(const bf16x2*)&cA[row * LDP + kk];
                af[mt][2 * j] = p[0]; af[mt][2 * j + 1] = p[1];
            }
        }
        for (int nt = 0; nt < 2; ++nt) {
            v16bf bfg;
            int col = wc * 32 + nt * 16 + ln;
            for (int j = 0; j < 8; ++j) {
                int kk = bkidx(j, half);
                bf16x2 p = *(const bf16x2*)&cB[col * LDP + kk];
                bfg[2 * j] = p[0]; bfg[2 * j + 1] = p[1];
            }
            for (int mt = 0; mt < 4; ++mt)
                acc[mt][nt] = wmma_bf16(af[mt], bfg, acc[mt][nt]);
        }
        __syncthreads();   // protect buffer reuse before next issueTile
    }

    // Epilogue (template-specialized; no runtime branches -> no spills).
    for (int mt = 0; mt < 4; ++mt) {
        for (int nt = 0; nt < 2; ++nt) {
            v8f v = acc[mt][nt];
            int n_g = blockN + wc * 32 + nt * 16 + ln;
            if constexpr (ROPE) {
                int p = (n_g & (HDc - 1)) >> 1;
                for (int r = 0; r < 8; ++r) {
                    int m_g = blockM + wr * 64 + mt * 16 + r + 8 * half;
                    int s   = m_g & (Sc - 1);
                    float cth = cosT[s * (HDc / 2) + p];
                    float sth = sinT[s * (HDc / 2) + p];
                    float a  = v[r];
                    float pv = __shfl_xor(a, 1, 32);
                    v[r] = (n_g & 1) ? (pv * sth + a * cth)   // imag lane
                                     : (a * cth - pv * sth);  // real lane
                }
            }
            for (int r = 0; r < 8; ++r) {
                int m_g = blockM + wr * 64 + mt * 16 + r + 8 * half;
                if constexpr (STOREMODE == 1) {
                    ((float*)Cout)[(size_t)m_g * N + n_g] = v[r];
                } else if constexpr (STOREMODE == 2) {
                    int bb = m_g >> 11, s = m_g & (Sc - 1);
                    int g = n_g >> 7,   d = n_g & (HDc - 1);
                    size_t idx = (((size_t)bb * KVc + g) * HDc + d) * Sc + s;
                    ((bf16*)Cout)[idx] = (bf16)v[r];
                } else {
                    ((bf16*)Cout)[(size_t)m_g * N + n_g] = (bf16)v[r];
                }
            }
        }
    }
}

// ---------------------------------------------------------------------------
// Flash attention: grid (S/128, H, B), 256 threads = 8 independent waves.
// Each wave owns 16 query rows; online softmax over causal key blocks of 32.
// K is [b][s][g*HD+d] (per-lane d contiguous); V is [b][g][d][s] (per-lane
// key contiguous) -> all K/V fragment loads vectorize to b128.
// ---------------------------------------------------------------------------
__launch_bounds__(256)
__global__ void flash_attn_wmma_k(const bf16* __restrict__ qb, const bf16* __restrict__ kb,
                                  const bf16* __restrict__ vt, bf16* __restrict__ ob)
{
    __shared__ bf16 pStage[8][16 * 32];   // per-wave P re-layout (C -> A frag)

    const int t    = threadIdx.x;
    const int lane = t & 31;
    const int w    = t >> 5;
    const int half = lane >> 4;
    const int ln   = lane & 15;
    const int b = blockIdx.z, h = blockIdx.y, g = h / REPc;
    const int qBase = blockIdx.x * 128 + w * 16;

    // Q fragments (16 rows x 128 head-dim) held in registers for the kernel.
    v16bf qf[4];
    {
        const size_t qrow0 = ((size_t)(b * Sc + qBase)) * NQc + (size_t)h * HDc;
        for (int c = 0; c < 4; ++c)
            for (int j = 0; j < 8; ++j) {
                int d = c * 32 + akidx(j, half);
                bf16x2 p = *(const bf16x2*)&qb[qrow0 + (size_t)ln * NQc + d];
                qf[c][2 * j] = p[0]; qf[c][2 * j + 1] = p[1];
            }
    }

    v8f o[8];
    for (int nc = 0; nc < 8; ++nc) o[nc] = {};
    float mst[8], lst[8];
    for (int r = 0; r < 8; ++r) { mst[r] = -3.0e38f; lst[r] = 0.0f; }

    const size_t kvbase  = (size_t)b * Sc * NKVc + (size_t)g * HDc;       // K rows
    const size_t vtbase  = ((size_t)b * KVc + g) * HDc * Sc;              // V cols
    const int nkb = (qBase + 16 + 31) >> 5;   // causal: keys 0 .. qBase+15

    for (int kbI = 0; kbI < nkb; ++kbI) {
        const int keyBase = kbI * 32;

        // Prefetch next key block's K rows into GL2 while this one computes.
        if (kbI + 1 < nkb) {
            int nkey = keyBase + 32 + lane;
            __builtin_prefetch(&kb[kvbase + (size_t)nkey * NKVc], 0, 0);
        }

        // ---- scores: S(16x32) = Q(16x128) * K^T(128x32), SCALE + causal mask
        v8f sc[2];
        sc[0] = {}; sc[1] = {};
        for (int tt = 0; tt < 2; ++tt) {
            int key = keyBase + tt * 16 + ln;
            size_t krow = kvbase + (size_t)key * NKVc;
            for (int c = 0; c < 4; ++c) {
                v16bf bk;
                for (int j = 0; j < 8; ++j) {
                    int d = c * 32 + bkidx(j, half);
                    bf16x2 p = *(const bf16x2*)&kb[krow + d];
                    bk[2 * j] = p[0]; bk[2 * j + 1] = p[1];
                }
                sc[tt] = wmma_bf16(qf[c], bk, sc[tt]);
            }
        }
        for (int tt = 0; tt < 2; ++tt) {
            int key = keyBase + tt * 16 + ln;
            for (int r = 0; r < 8; ++r) {
                int qg = qBase + r + 8 * half;
                float x = sc[tt][r] * SCALEc;
                sc[tt][r] = (key <= qg) ? x : -3.0e38f;
            }
        }

        // ---- online softmax (row stats live per (r, half); 16-lane butterflies)
        for (int r = 0; r < 8; ++r) {
            float mx = fmaxf(sc[0][r], sc[1][r]);
            mx = fmaxf(mx, __shfl_xor(mx, 1, 32));
            mx = fmaxf(mx, __shfl_xor(mx, 2, 32));
            mx = fmaxf(mx, __shfl_xor(mx, 4, 32));
            mx = fmaxf(mx, __shfl_xor(mx, 8, 32));
            float mnew = fmaxf(mst[r], mx);
            float corr = __expf(mst[r] - mnew);
            mst[r] = mnew;
            float p0 = __expf(sc[0][r] - mnew);
            float p1 = __expf(sc[1][r] - mnew);
            float rs = p0 + p1;
            rs += __shfl_xor(rs, 1, 32);
            rs += __shfl_xor(rs, 2, 32);
            rs += __shfl_xor(rs, 4, 32);
            rs += __shfl_xor(rs, 8, 32);
            lst[r] = lst[r] * corr + rs;
            for (int nc = 0; nc < 8; ++nc) o[nc][r] *= corr;
            int m = r + 8 * half;
            pStage[w][m * 32 + ln]      = (bf16)p0;
            pStage[w][m * 32 + 16 + ln] = (bf16)p1;
        }

        // per-wave LDS RAW: in-order DS pipe; block compiler motion + wait
        __builtin_amdgcn_wave_barrier();
        asm volatile("s_wait_dscnt 0x0" ::: "memory");

        v16bf pa;
        for (int j = 0; j < 8; ++j) {
            int kk = akidx(j, half);
            bf16x2 p = *(const bf16x2*)&pStage[w][ln * 32 + kk];
            pa[2 * j] = p[0]; pa[2 * j + 1] = p[1];
        }
        __builtin_amdgcn_wave_barrier();
        asm volatile("" ::: "memory");

        // ---- O += P(16x32) * V(32x128)  (V transposed: key contiguous per lane)
        for (int nc = 0; nc < 8; ++nc) {
            v16bf bv;
            int vcol = nc * 16 + ln;
            size_t vrow = vtbase + (size_t)vcol * Sc;
            for (int j = 0; j < 8; ++j) {
                int key = keyBase + bkidx(j, half);
                bf16x2 p = *(const bf16x2*)&vt[vrow + key];
                bv[2 * j] = p[0]; bv[2 * j + 1] = p[1];
            }
            o[nc] = wmma_bf16(pa, bv, o[nc]);
        }
    }

    // ---- normalize + store bf16 [b, s, h*HD + d]
    float inv[8];
    for (int r = 0; r < 8; ++r) inv[r] = 1.0f / lst[r];
    for (int nc = 0; nc < 8; ++nc)
        for (int r = 0; r < 8; ++r) {
            int m_g = qBase + r + 8 * half;
            size_t idx = ((size_t)(b * Sc + m_g)) * NQc + (size_t)h * HDc + nc * 16 + ln;
            ob[idx] = (bf16)(o[nc][r] * inv[r]);
        }
}

// ---------------------------------------------------------------------------
// Host-side launcher
// ---------------------------------------------------------------------------
extern "C" void kernel_launch(void* const* d_in, const int* in_sizes, int n_in,
                              void* d_out, int out_size, void* d_ws, size_t ws_size,
                              hipStream_t stream)
{
    (void)in_sizes; (void)n_in; (void)out_size; (void)ws_size;
    const float* x    = (const float*)d_in[0];
    const float* wq   = (const float*)d_in[1];
    const float* wk   = (const float*)d_in[2];
    const float* wv   = (const float*)d_in[3];
    const float* wo   = (const float*)d_in[4];
    const float* cosT = (const float*)d_in[5];
    const float* sinT = (const float*)d_in[6];
    // d_in[7] (mask) implemented as causal predicate; d_in[8] (positions) is arange.

    char* ws = (char*)d_ws;
    size_t off = 0;
    bf16* xb    = (bf16*)(ws + off); off += (size_t)Mc * DIMc * 2;      // 32 MiB
    bf16* wqT   = (bf16*)(ws + off); off += (size_t)DIMc * NQc * 2;     // 32 MiB (NQ x DIM)
    bf16* wkT   = (bf16*)(ws + off); off += (size_t)DIMc * NKVc * 2;    //  8 MiB (NKV x DIM)
    bf16* wvT   = (bf16*)(ws + off); off += (size_t)DIMc * NKVc * 2;    //  8 MiB (NKV x DIM)
    bf16* woT   = (bf16*)(ws + off); off += (size_t)NQc * DIMc * 2;     // 32 MiB (DIM x NQ)
    bf16* qbuf  = (bf16*)(ws + off); off += (size_t)Mc * NQc * 2;       // 32 MiB
    bf16* kbuf  = (bf16*)(ws + off); off += (size_t)Mc * NKVc * 2;      //  8 MiB
    bf16* vtbuf = (bf16*)(ws + off); off += (size_t)Mc * NKVc * 2;      //  8 MiB [b][g][d][s]
    bf16* attnb = (bf16*)(ws + off); off += (size_t)Mc * NQc * 2;       // 32 MiB

    // Activations: straight conversion. Weights: transpose to N x K.
    cvt_bf16_k<<<2048, 256, 0, stream>>>(x, xb, Mc * DIMc);
    cvt_bf16_t_k<<<dim3(NQc / 32,  DIMc / 32), 256, 0, stream>>>(wq, wqT, DIMc, NQc);
    cvt_bf16_t_k<<<dim3(NKVc / 32, DIMc / 32), 256, 0, stream>>>(wk, wkT, DIMc, NKVc);
    cvt_bf16_t_k<<<dim3(NKVc / 32, DIMc / 32), 256, 0, stream>>>(wv, wvT, DIMc, NKVc);
    cvt_bf16_t_k<<<dim3(DIMc / 32, NQc / 32),  256, 0, stream>>>(wo, woT, NQc, DIMc);

    // Q = x@wq (+RoPE), K = x@wk (+RoPE), V = x@wv (stored transposed)
    gemm_bf16_wmma_k<1, 0><<<dim3(NQc / 128,  Mc / 128), 256, 0, stream>>>(
        xb, wqT, qbuf, Mc, NQc, DIMc, cosT, sinT);
    gemm_bf16_wmma_k<1, 0><<<dim3(NKVc / 128, Mc / 128), 256, 0, stream>>>(
        xb, wkT, kbuf, Mc, NKVc, DIMc, cosT, sinT);
    gemm_bf16_wmma_k<0, 2><<<dim3(NKVc / 128, Mc / 128), 256, 0, stream>>>(
        xb, wvT, vtbuf, Mc, NKVc, DIMc, nullptr, nullptr);

    // causal grouped flash attention
    flash_attn_wmma_k<<<dim3(Sc / 128, Hc, Bc), 256, 0, stream>>>(
        qbuf, kbuf, vtbuf, attnb);

    // out = attn @ wo  (f32 straight to d_out)
    gemm_bf16_wmma_k<0, 1><<<dim3(DIMc / 128, Mc / 128), 256, 0, stream>>>(
        attnb, woT, (void*)d_out, Mc, DIMc, NQc, nullptr, nullptr);
}
